// TransformerBlock_29652454211679
// MI455X (gfx1250) — matrix operand order repaired and checked by reading, not compiled, not run on hardware
//
#include <hip/hip_runtime.h>
#include <math.h>

// ---------------------------------------------------------------------------
// TransformerBlock for MI455X (gfx1250, wave32, WMMA 16x16x32 f16->f32).
//
//   1. x  -> f16            (Xh)
//   2. W* -> f16 transposed (N x K layout so B-fragments are K-contiguous)
//   3. Q/K/V = Xh @ W*T + b (LDS-staged WMMA GEMM, f16 out)
//   4. V  -> per-(n,h) transpose (64 x 1024) for the O-GEMM B operand
//   5. fused attention: scores(WMMA) -> softmax over HEADS (LDS) -> O(WMMA)
//   6. proj = O @ WoT + bo  (fp32 out)
//   7. x1 = LN(proj + x)    (fp32 + f16 out)
//   8. H1 = gelu(x1 @ W1T + c1)
//   9. F  = H1 @ W2T + c2
//  10. out = LN(F + x1)     (fp32, d_out)
//
// GEMM: block tile 256x128, 8 waves of 64x64 (16 WMMA / K-step), K-step 32,
// double-buffered LDS staging via GLOBAL_LOAD_ASYNC_TO_LDS_B128
// (ASYNCcnt + s_wait_asynccnt), with load+ds_store fallback.
// ---------------------------------------------------------------------------

typedef __attribute__((ext_vector_type(16))) _Float16 v16h;
typedef __attribute__((ext_vector_type(8)))  _Float16 v8h;
typedef __attribute__((ext_vector_type(8)))  float    v8f;

#define WMMA16x16x32(a, b, c) \
  __builtin_amdgcn_wmma_f32_16x16x32_f16(false, (a), false, (b), (short)0, (c), false, false)

#if defined(__HIP_DEVICE_COMPILE__) && __has_builtin(__builtin_amdgcn_global_load_async_to_lds_b128)
#define USE_ASYNC_LDS 1
#else
#define USE_ASYNC_LDS 0
#endif

#if USE_ASYNC_LDS
// Builtin signature (from hipcc diagnostic): (int4 AS1* gsrc, int4 AS3* ldst,
// Imm offset, Imm cpol). AS(1) prints as __device__ in HIP.
typedef int vi4 __attribute__((vector_size(16)));
typedef __attribute__((address_space(1))) vi4 gvi4_t;
typedef __attribute__((address_space(3))) vi4 lvi4_t;
#endif

// Copy 16B global -> LDS. Async path uses the CDNA5 async LDS pipe (ASYNCcnt);
// low 32 bits of a flat shared pointer are the LDS offset (ISA 10.2).
__device__ __forceinline__ void cp_b128(_Float16* lds_dst, const _Float16* gsrc) {
#if USE_ASYNC_LDS
  __builtin_amdgcn_global_load_async_to_lds_b128(
      (gvi4_t*)(uintptr_t)gsrc,
      (lvi4_t*)(unsigned int)(uintptr_t)lds_dst, 0, 0);
#else
  *(v8h*)lds_dst = *(const v8h*)gsrc;
#endif
}

__device__ __forceinline__ void wait_async_tail6() {  // allow the 6 just-issued
#if USE_ASYNC_LDS
#if __has_builtin(__builtin_amdgcn_s_wait_asynccnt)
  __builtin_amdgcn_s_wait_asynccnt(6);
#else
  asm volatile("s_wait_asynccnt 0x6" ::: "memory");
#endif
#endif
}
__device__ __forceinline__ void wait_async_all() {
#if USE_ASYNC_LDS
#if __has_builtin(__builtin_amdgcn_s_wait_asynccnt)
  __builtin_amdgcn_s_wait_asynccnt(0);
#else
  asm volatile("s_wait_asynccnt 0x0" ::: "memory");
#endif
#endif
}

// A-matrix fragment (16x32, f16). ISA 7.12.2: lane holds row M=lane%16,
// K chunks {hi*8..+7} and {16+hi*8..+7}  (hi = lane/16).
__device__ __forceinline__ v16h frag_a(const _Float16* __restrict__ p, int ld) {
  int lane = threadIdx.x & 31;
  int row = lane & 15, hi = lane >> 4;
  const _Float16* q = p + (size_t)row * ld + hi * 8;
  v8h lo = *(const v8h*)q;
  v8h hv = *(const v8h*)(q + 16);
  return __builtin_shufflevector(lo, hv, 0, 1, 2, 3, 4, 5, 6, 7, 8, 9, 10, 11, 12, 13, 14, 15);
}

// B-matrix fragment (32x16) from B^T stored (N x K) row-major.
// Lane holds N = lane%16, K chunks {hi*8..+7} and {16+hi*8..+7} (hi = lane/16).
__device__ __forceinline__ v16h frag_b(const _Float16* __restrict__ p, int ld) {
  int lane = threadIdx.x & 31;
  int row = lane & 15, hi = lane >> 4;
  const _Float16* q = p + (size_t)row * ld + hi * 8;
  v8h lo = *(const v8h*)q;
  v8h hv = *(const v8h*)(q + 16);
  return __builtin_shufflevector(lo, hv, 0, 1, 2, 3, 4, 5, 6, 7, 8, 9, 10, 11, 12, 13, 14, 15);
}

// ---------------------------------------------------------------------------
__global__ void __launch_bounds__(256) cvt16_kernel(const float* __restrict__ X,
                                                    _Float16* __restrict__ Y, int n) {
  int i = (blockIdx.x * 256 + threadIdx.x) * 4;
  if (i + 3 < n) {
#pragma unroll
    for (int j = 0; j < 4; ++j) Y[i + j] = (_Float16)X[i + j];
  }
}

// fp32 (R x C) -> fp16 transposed (C x R); grid = (C/32, R/32), block = 256
__global__ void __launch_bounds__(256) transpose_cvt_kernel(const float* __restrict__ W,
                                                            _Float16* __restrict__ Wt,
                                                            int R, int C) {
  __shared__ _Float16 tile[32][33];
  int c0 = blockIdx.x * 32, r0 = blockIdx.y * 32;
  int tx = threadIdx.x & 31, ty = threadIdx.x >> 5;
#pragma unroll
  for (int i = 0; i < 32; i += 8)
    tile[ty + i][tx] = (_Float16)W[(size_t)(r0 + ty + i) * C + c0 + tx];
  __syncthreads();
#pragma unroll
  for (int i = 0; i < 32; i += 8)
    Wt[(size_t)(c0 + ty + i) * R + r0 + tx] = tile[tx][ty + i];
}

// Per-(n,h) transpose of V: (1024 s x 64 d) slab -> (64 d x 1024 s).
__global__ void __launch_bounds__(256) vtrans_kernel(const _Float16* __restrict__ V,
                                                     _Float16* __restrict__ Vt) {
  __shared__ _Float16 t[64][65];
  int slab = blockIdx.y, s0 = blockIdx.x * 64;
  const _Float16* vb = V + (size_t)slab * 65536;
  _Float16* ob = Vt + (size_t)slab * 65536;
  int tx = threadIdx.x & 63, ty = threadIdx.x >> 6;
#pragma unroll
  for (int i = 0; i < 64; i += 4) t[ty + i][tx] = vb[(size_t)(s0 + ty + i) * 64 + tx];
  __syncthreads();
#pragma unroll
  for (int i = 0; i < 64; i += 4) ob[(size_t)(ty + i) * 1024 + s0 + tx] = t[tx][ty + i];
}

// ---------------------------------------------------------------------------
// LDS-staged WMMA GEMM: C[M x N] = A[M x K](f16) @ Bt[N x K](f16)^T + bias.
// grid = (N/128, M/256), block = 256 = 8 waves (4 M-waves x 2 N-waves).
template <int ACT>
__global__ void __launch_bounds__(256) gemm_wmma_kernel(
    const _Float16* __restrict__ A, const _Float16* __restrict__ Bt,
    const float* __restrict__ bias, float* __restrict__ C32,
    _Float16* __restrict__ C16, int M, int N, int K) {
  // LDS: rows padded to 40 halfs (80B) to spread fragment reads over banks.
  __shared__ _Float16 As[2][256][40];  // 40 KB
  __shared__ _Float16 Bs[2][128][40];  // 20 KB
  int tid = threadIdx.x;
  int wave = tid >> 5, lane = tid & 31;
  int wm = wave & 3, wn = wave >> 2;
  int mBlk = blockIdx.y * 256, nBlk = blockIdx.x * 128;
  int mW = wm * 64, nW = wn * 64;

  const _Float16* Arow = A + (size_t)(mBlk + tid) * K;  // this thread's A row
  int brow = tid & 127, bsel = tid >> 7;                // B row + half-row select
  const _Float16* Brow = Bt + (size_t)(nBlk + brow) * K + bsel * 16;

  const v8f z = {0.f, 0.f, 0.f, 0.f, 0.f, 0.f, 0.f, 0.f};
  v8f acc[4][4];
#pragma unroll
  for (int i = 0; i < 4; ++i)
#pragma unroll
    for (int j = 0; j < 4; ++j) acc[i][j] = z;

  // stage one 32-wide K chunk into buffer `buf` (6 x b128 per thread)
  auto stage = [&](int k, int buf) {
#pragma unroll
    for (int c = 0; c < 4; ++c) cp_b128(&As[buf][tid][c * 8], Arow + k + c * 8);
#pragma unroll
    for (int c = 0; c < 2; ++c)
      cp_b128(&Bs[buf][brow][bsel * 16 + c * 8], Brow + k + c * 8);
  };

  int nk = K >> 5;
  stage(0, 0);
  for (int kt = 0; kt < nk; ++kt) {
    int buf = kt & 1;
    if (kt + 1 < nk) {
      stage((kt + 1) << 5, buf ^ 1);
      wait_async_tail6();  // current buffer's 6 copies done; next 6 in flight
    } else {
      wait_async_all();
    }
    __syncthreads();

    v16h af[4], bf[4];
#pragma unroll
    for (int i = 0; i < 4; ++i) af[i] = frag_a(&As[buf][mW + i * 16][0], 40);
#pragma unroll
    for (int j = 0; j < 4; ++j) bf[j] = frag_b(&Bs[buf][nW + j * 16][0], 40);
#pragma unroll
    for (int i = 0; i < 4; ++i)
#pragma unroll
      for (int j = 0; j < 4; ++j) acc[i][j] = WMMA16x16x32(af[i], bf[j], acc[i][j]);

    __syncthreads();  // protect `buf` (re-staged next iteration) from laggards
  }

  int col16 = lane & 15, rhi = (lane >> 4) * 8;
#pragma unroll
  for (int i = 0; i < 4; ++i)
#pragma unroll
    for (int j = 0; j < 4; ++j) {
      int col = nBlk + nW + j * 16 + col16;
      float bv = bias ? bias[col] : 0.f;
#pragma unroll
      for (int r = 0; r < 8; ++r) {
        int row = mBlk + mW + i * 16 + rhi + r;
        float v = acc[i][j][r] + bv;
        if (ACT == 1) v = 0.5f * v * (1.0f + erff(v * 0.70710678118654752f));
        if (C32) C32[(size_t)row * N + col] = v;
        if (C16) C16[(size_t)row * N + col] = (_Float16)v;
      }
    }
}

// ---------------------------------------------------------------------------
// Fused attention; softmax over the HEAD axis (reference dim=1) is local to
// each (n,t,s), so a block owns 16 t-rows x all 16 heads and loops s-tiles.
__global__ void __launch_bounds__(256) attn_kernel(const _Float16* __restrict__ Q,
                                                   const _Float16* __restrict__ Kb,
                                                   const _Float16* __restrict__ Vt,
                                                   _Float16* __restrict__ O) {
  __shared__ float S[16][16][32];       // [h][t][s] 32 KB
  __shared__ _Float16 Aat[16][16][32];  // 16 KB
  int n = blockIdx.y, t0 = blockIdx.x * 16;
  int wave = threadIdx.x >> 5, lane = threadIdx.x & 31;
  int col16 = lane & 15, rhi = (lane >> 4) * 8;
  const v8f z = {0.f, 0.f, 0.f, 0.f, 0.f, 0.f, 0.f, 0.f};

  v16h aq[2][2];
#pragma unroll
  for (int p = 0; p < 2; ++p) {
    int h = wave * 2 + p;
    const _Float16* qb = Q + ((size_t)(n * 16 + h) * 1024 + t0) * 64;
    aq[p][0] = frag_a(qb, 64);
    aq[p][1] = frag_a(qb + 32, 64);
  }

  v8f accO[2][4];
#pragma unroll
  for (int p = 0; p < 2; ++p)
#pragma unroll
    for (int nt = 0; nt < 4; ++nt) accO[p][nt] = z;

  for (int s0 = 0; s0 < 1024; s0 += 32) {
#pragma unroll
    for (int p = 0; p < 2; ++p) {
      int h = wave * 2 + p;
      const _Float16* kb = Kb + ((size_t)(n * 16 + h) * 1024 + s0) * 64;
#pragma unroll
      for (int nt = 0; nt < 2; ++nt) {
        v16h b0 = frag_b(kb + nt * 16 * 64, 64);
        v16h b1 = frag_b(kb + nt * 16 * 64 + 32, 64);
        v8f s = z;
        s = WMMA16x16x32(aq[p][0], b0, s);
        s = WMMA16x16x32(aq[p][1], b1, s);
#pragma unroll
        for (int r = 0; r < 8; ++r) S[h][rhi + r][nt * 16 + col16] = s[r];
      }
    }
    __syncthreads();

#pragma unroll
    for (int pp = 0; pp < 2; ++pp) {
      int pos = threadIdx.x + pp * 256;
      int t = pos >> 5, ss = pos & 31;
      float vals[16], mx = -3.4e38f;
#pragma unroll
      for (int h = 0; h < 16; ++h) {
        float v = S[h][t][ss] * 0.03125f;  // 1/sqrt(1024)
        vals[h] = v;
        mx = fmaxf(mx, v);
      }
      float sum = 0.f;
#pragma unroll
      for (int h = 0; h < 16; ++h) {
        float e = expf(vals[h] - mx);
        vals[h] = e;
        sum += e;
      }
      float inv = 1.0f / sum;
#pragma unroll
      for (int h = 0; h < 16; ++h) Aat[h][t][ss] = (_Float16)(vals[h] * inv);
    }
    __syncthreads();

#pragma unroll
    for (int p = 0; p < 2; ++p) {
      int h = wave * 2 + p;
      v16h aA = frag_a(&Aat[h][0][0], 32);
      const _Float16* vb = Vt + (size_t)(n * 16 + h) * 65536 + s0;
#pragma unroll
      for (int nt = 0; nt < 4; ++nt) {
        v16h bV = frag_b(vb + (size_t)nt * 16 * 1024, 1024);
        accO[p][nt] = WMMA16x16x32(aA, bV, accO[p][nt]);
      }
    }
    __syncthreads();
  }

#pragma unroll
  for (int p = 0; p < 2; ++p) {
    int h = wave * 2 + p;
#pragma unroll
    for (int nt = 0; nt < 4; ++nt) {
      int d = nt * 16 + col16;
#pragma unroll
      for (int r = 0; r < 8; ++r) {
        int t = t0 + rhi + r;
        O[((size_t)(n * 16 + h) * 1024 + t) * 64 + d] = (_Float16)accO[p][nt][r];
      }
    }
  }
}

// ---------------------------------------------------------------------------
__global__ void __launch_bounds__(256) ln_kernel(const float* __restrict__ Y,
                                                 const float* __restrict__ R,
                                                 const float* __restrict__ g,
                                                 const float* __restrict__ b,
                                                 float* __restrict__ out32,
                                                 _Float16* __restrict__ out16) {
  int row = blockIdx.x;
  const float* y = Y + (size_t)row * 1024;
  const float* r = R + (size_t)row * 1024;
  float v[4], s = 0.f, s2 = 0.f;
#pragma unroll
  for (int i = 0; i < 4; ++i) {
    int idx = threadIdx.x + i * 256;
    float t = y[idx] + r[idx];
    v[i] = t;
    s += t;
    s2 += t * t;
  }
#pragma unroll
  for (int off = 16; off > 0; off >>= 1) {
    s += __shfl_xor(s, off, 32);
    s2 += __shfl_xor(s2, off, 32);
  }
  __shared__ float ws1[8], ws2[8], mv[2];
  int wave = threadIdx.x >> 5, lane = threadIdx.x & 31;
  if (lane == 0) { ws1[wave] = s; ws2[wave] = s2; }
  __syncthreads();
  if (threadIdx.x == 0) {
    float a = 0.f, c = 0.f;
#pragma unroll
    for (int i = 0; i < 8; ++i) { a += ws1[i]; c += ws2[i]; }
    float mean = a * (1.0f / 1024.0f);
    float var = c * (1.0f / 1024.0f) - mean * mean;
    mv[0] = mean;
    mv[1] = rsqrtf(var + 1e-5f);
  }
  __syncthreads();
  float mean = mv[0], inv = mv[1];
#pragma unroll
  for (int i = 0; i < 4; ++i) {
    int idx = threadIdx.x + i * 256;
    float o = (v[i] - mean) * inv * g[idx] + b[idx];
    if (out32) out32[(size_t)row * 1024 + idx] = o;
    if (out16) out16[(size_t)row * 1024 + idx] = (_Float16)o;
  }
}

// ---------------------------------------------------------------------------
extern "C" void kernel_launch(void* const* d_in, const int* in_sizes, int n_in,
                              void* d_out, int out_size, void* d_ws, size_t ws_size,
                              hipStream_t stream) {
  (void)in_sizes; (void)n_in; (void)out_size; (void)ws_size;
  const float* x  = (const float*)d_in[0];
  const float* Wq = (const float*)d_in[1];
  const float* bq = (const float*)d_in[2];
  const float* Wk = (const float*)d_in[3];
  const float* bk = (const float*)d_in[4];
  const float* Wv = (const float*)d_in[5];
  const float* bv = (const float*)d_in[6];
  const float* Wo = (const float*)d_in[7];
  const float* bo = (const float*)d_in[8];
  const float* g1 = (const float*)d_in[9];
  const float* b1 = (const float*)d_in[10];
  const float* W1 = (const float*)d_in[11];
  const float* c1 = (const float*)d_in[12];
  const float* W2 = (const float*)d_in[13];
  const float* c2 = (const float*)d_in[14];
  const float* g2 = (const float*)d_in[15];
  const float* b2 = (const float*)d_in[16];

  char* ws = (char*)d_ws;
  size_t off = 0;
  auto take = [&](size_t bytes) -> char* {
    char* p = ws + off;
    off += (bytes + 255) & ~(size_t)255;
    return p;
  };
  const size_t M = 4096, H = 1024, F = 4096;
  _Float16* Xh  = (_Float16*)take(M * H * 2);
  _Float16* WqT = (_Float16*)take(H * H * 2);
  _Float16* WkT = (_Float16*)take(H * H * 2);
  _Float16* WvT = (_Float16*)take(H * H * 2);
  _Float16* WoT = (_Float16*)take(H * H * 2);
  _Float16* W1T = (_Float16*)take(F * H * 2);
  _Float16* W2T = (_Float16*)take(H * F * 2);
  _Float16* Qb  = (_Float16*)take(M * H * 2);
  _Float16* Kbf = (_Float16*)take(M * H * 2);
  _Float16* Vb  = (_Float16*)take(M * H * 2);
  _Float16* Vt  = (_Float16*)take(M * H * 2);
  _Float16* Ob  = (_Float16*)take(M * H * 2);
  float*    P32 = (float*)take(M * H * 4);
  float*    X1f = (float*)take(M * H * 4);
  _Float16* X1h = (_Float16*)take(M * H * 2);
  _Float16* H1  = (_Float16*)take(M * F * 2);

  cvt16_kernel<<<4096, 256, 0, stream>>>(x, Xh, (int)(M * H));
  transpose_cvt_kernel<<<dim3(32, 32), 256, 0, stream>>>(Wq, WqT, 1024, 1024);
  transpose_cvt_kernel<<<dim3(32, 32), 256, 0, stream>>>(Wk, WkT, 1024, 1024);
  transpose_cvt_kernel<<<dim3(32, 32), 256, 0, stream>>>(Wv, WvT, 1024, 1024);
  transpose_cvt_kernel<<<dim3(32, 32), 256, 0, stream>>>(Wo, WoT, 1024, 1024);
  transpose_cvt_kernel<<<dim3(128, 32), 256, 0, stream>>>(W1, W1T, 1024, 4096);
  transpose_cvt_kernel<<<dim3(32, 128), 256, 0, stream>>>(W2, W2T, 4096, 1024);
  gemm_wmma_kernel<0><<<dim3(8, 16), 256, 0, stream>>>(Xh, WqT, bq, nullptr, Qb, 4096, 1024, 1024);
  gemm_wmma_kernel<0><<<dim3(8, 16), 256, 0, stream>>>(Xh, WkT, bk, nullptr, Kbf, 4096, 1024, 1024);
  gemm_wmma_kernel<0><<<dim3(8, 16), 256, 0, stream>>>(Xh, WvT, bv, nullptr, Vb, 4096, 1024, 1024);
  vtrans_kernel<<<dim3(16, 64), 256, 0, stream>>>(Vb, Vt);
  attn_kernel<<<dim3(64, 4), 256, 0, stream>>>(Qb, Kbf, Vt, Ob);
  gemm_wmma_kernel<0><<<dim3(8, 16), 256, 0, stream>>>(Ob, WoT, bo, P32, nullptr, 4096, 1024, 1024);
  ln_kernel<<<4096, 256, 0, stream>>>(P32, x, g1, b1, X1f, X1h);
  gemm_wmma_kernel<1><<<dim3(32, 16), 256, 0, stream>>>(X1h, W1T, c1, nullptr, H1, 4096, 4096, 1024);
  gemm_wmma_kernel<0><<<dim3(8, 16), 256, 0, stream>>>(H1, W2T, c2, P32, nullptr, 4096, 1024, 4096);
  ln_kernel<<<4096, 256, 0, stream>>>(P32, X1f, g2, b2, (float*)d_out, nullptr);
}
